// MutualCrossAttention_62431644614923
// MI455X (gfx1250) — matrix operand order repaired and checked by reading, not compiled, not run on hardware
//
#include <hip/hip_runtime.h>

// Mutual cross attention for MI455X (gfx1250, wave32, WMMA bf16).
// out = attn(x1,x2) + attn(x2,x1), attn(q,k) = softmax(q k^T / sqrt(D)) @ k
// Flash-attention style, async global->LDS staging of K blocks (ASYNCcnt)
// overlapped with WMMA compute, software-pipelined LDS operand fetch.

typedef __bf16 bf16_t;
typedef __attribute__((ext_vector_type(8)))  __bf16 v8bf;
typedef __attribute__((ext_vector_type(16))) __bf16 v16bf;
typedef __attribute__((ext_vector_type(8)))  float  v8f;

#define BM       128   // query rows per workgroup (16 per wave)
#define BN       64    // key rows per inner iteration
#define DD       256   // feature dim
#define NSEQ     2048
#define NWAVE    8
#define NTHREADS 256
#define NKB      (NSEQ / BN)
#define LOG2E    1.44269504088896f

__device__ __forceinline__ v8f v8f_zero() {
    v8f z = {0.f, 0.f, 0.f, 0.f, 0.f, 0.f, 0.f, 0.f};
    return z;
}

// Async copy of 16 bytes global -> LDS (CDNA5, tracked by ASYNCcnt).
__device__ __forceinline__ void async_ld_b128(void* lds, const float* g) {
    const unsigned lds_off = (unsigned)(unsigned long long)lds; // addr[31:0] = LDS offset
    asm volatile("global_load_async_to_lds_b128 %0, %1, off"
                 :: "v"(lds_off), "v"(g)
                 : "memory");
}
__device__ __forceinline__ void wait_async0() {
    asm volatile("s_wait_asynccnt 0x0" ::: "memory");
}

// A-matrix (16x32 bf16) per-lane packing (ISA 7.12.2):
// lanes 0-15 : K = koff+0..7  and koff+16..23
// lanes 16-31: K = koff+8..15 and koff+24..31
__device__ __forceinline__ v16bf loadA(const bf16_t* row, int koff, int hi16) {
    const v8bf c0 = *(const v8bf*)(row + koff + hi16 * 8);
    const v8bf c1 = *(const v8bf*)(row + koff + 16 + hi16 * 8);
    v16bf r;
#pragma unroll
    for (int i = 0; i < 8; ++i) { r[i] = c0[i]; r[i + 8] = c1[i]; }
    return r;
}

// B-matrix (32x16 bf16) per-lane packing: column n = lane%16,
// lanes 0-15 hold K = koff+0..7 and koff+16..23, lanes 16-31 hold K = koff+8..15 and koff+24..31.
__device__ __forceinline__ v16bf loadB(const bf16_t* row, int koff, int hi16) {
    const v8bf c0 = *(const v8bf*)(row + koff + hi16 * 8);
    const v8bf c1 = *(const v8bf*)(row + koff + 16 + hi16 * 8);
    v16bf r;
#pragma unroll
    for (int i = 0; i < 8; ++i) { r[i] = c0[i]; r[i + 8] = c1[i]; }
    return r;
}

__global__ __launch_bounds__(NTHREADS, 1)
void mca_flash_attn(const float* __restrict__ Q, const float* __restrict__ K,
                    float* __restrict__ Out, int accumulate)
{
    extern __shared__ char smem_raw[];
    bf16_t* sQ   = (bf16_t*)smem_raw;       // [BM][DD]  64 KB (pre-scaled bf16)
    bf16_t* sK   = sQ  + BM * DD;           // [BN][DD]  32 KB (row major)
    bf16_t* sKt  = sK  + BN * DD;           // [DD][BN]  32 KB (transposed)
    bf16_t* sP   = sKt + DD * BN;           // [NWAVE][16][BN] 16 KB probs
    float*  sRaw = (float*)(sP + NWAVE * 16 * BN); // [BN][DD] fp32, 64 KB staging

    const int tid  = threadIdx.x;
    const int wave = tid >> 5;
    const int lane = tid & 31;
    const int hi16 = lane >> 4;
    const int l16  = lane & 15;

    const int b    = blockIdx.y;
    const int row0 = blockIdx.x * BM;

    const float* Qg = Q   + ((size_t)b * NSEQ + row0) * DD;
    const float* Kg = K   + (size_t)b * NSEQ * DD;
    float*       Og = Out + ((size_t)b * NSEQ + row0) * DD;

    // Fold 1/sqrt(D) into Q at conversion time.
    const float scale = 0.0625f;
#pragma unroll 4
    for (int i = tid; i < BM * DD / 4; i += NTHREADS) {
        float4 v = ((const float4*)Qg)[i];
        bf16_t* d = sQ + i * 4;
        d[0] = (bf16_t)(v.x * scale); d[1] = (bf16_t)(v.y * scale);
        d[2] = (bf16_t)(v.z * scale); d[3] = (bf16_t)(v.w * scale);
    }

    // Kick off async staging of K block 0 (raw fp32 -> LDS, 16 x b128 per lane).
#pragma unroll
    for (int j = 0; j < BN * DD / 4 / NTHREADS; ++j) {
        const int i = tid + j * NTHREADS;           // float4 index
        async_ld_b128(sRaw + i * 4, Kg + (size_t)i * 4);
    }

    // Accumulators: O is 16 rows x 256 cols (16 C-tiles), flash stats per row.
    v8f o[16];
#pragma unroll
    for (int t = 0; t < 16; ++t) o[t] = v8f_zero();
    float m_r[8], l_r[8];
#pragma unroll
    for (int r = 0; r < 8; ++r) { m_r[r] = -3.0e38f; l_r[r] = 0.0f; }

    const bf16_t* qrow = sQ + (wave * 16 + l16) * DD;
    bf16_t*       sPw  = sP + wave * 16 * BN;

    for (int kb = 0; kb < NKB; ++kb) {
        wait_async0();      // this wave's staged writes to sRaw have landed
        __syncthreads();    // all waves landed; previous compute readers done

        // Convert staged fp32 K block -> bf16 sK (row major) + sKt (transposed).
#pragma unroll 4
        for (int i = tid; i < BN * DD / 4; i += NTHREADS) {
            float4 v = ((const float4*)sRaw)[i];
            const int flat = i * 4;
            const int row  = flat >> 8;        // / DD
            const int col  = flat & (DD - 1);
            bf16_t b0 = (bf16_t)v.x, b1 = (bf16_t)v.y;
            bf16_t b2 = (bf16_t)v.z, b3 = (bf16_t)v.w;
            bf16_t* d = sK + row * DD + col;
            d[0] = b0; d[1] = b1; d[2] = b2; d[3] = b3;
            sKt[(col + 0) * BN + row] = b0;
            sKt[(col + 1) * BN + row] = b1;
            sKt[(col + 2) * BN + row] = b2;
            sKt[(col + 3) * BN + row] = b3;
        }
        __syncthreads();    // sK/sKt visible; sRaw free to overwrite

        // Stage next K block asynchronously; it overlaps all compute below.
        if (kb + 1 < NKB) {
            const float* Knext = Kg + (size_t)(kb + 1) * BN * DD;
#pragma unroll
            for (int j = 0; j < BN * DD / 4 / NTHREADS; ++j) {
                const int i = tid + j * NTHREADS;
                async_ld_b128(sRaw + i * 4, Knext + (size_t)i * 4);
            }
            if (kb + 2 < NKB)   // pull block kb+2 HBM -> L2 ahead of time
                __builtin_prefetch(Kg + (size_t)(kb + 2) * BN * DD + tid * 64, 0, 1);
        }

        // ---- GEMM 1: S (16 x BN) = Qw (16 x D) * K^T ----
        // Flattened (kt x nt) loop, operands fetched one step ahead so each
        // WMMA overlaps the next tile's ds_load latency.
        v8f s[4];
#pragma unroll
        for (int t = 0; t < 4; ++t) s[t] = v8f_zero();
        {
            v16bf a_cur = loadA(qrow, 0, hi16);
            v16bf b_cur = loadB(sK + l16 * DD, 0, hi16);
#pragma unroll
            for (int idx = 0; idx < (DD / 32) * (BN / 16); ++idx) {
                const int nt = idx & 3;
                v16bf a_nxt = a_cur;
                v16bf b_nxt = b_cur;
                if (idx + 1 < (DD / 32) * (BN / 16)) {
                    const int nkt = (idx + 1) >> 2;
                    const int nnt = (idx + 1) & 3;
                    if (nnt == 0) a_nxt = loadA(qrow, nkt * 32, hi16);
                    b_nxt = loadB(sK + (nnt * 16 + l16) * DD, nkt * 32, hi16);
                }
                s[nt] = __builtin_amdgcn_wmma_f32_16x16x32_bf16(
                    false, a_cur, false, b_cur, (short)0, s[nt], false, false);
                a_cur = a_nxt;
                b_cur = b_nxt;
            }
        }

        // ---- online softmax (C-layout: row m = r + 8*hi16, col n = l16) ----
        float alpha[8];
#pragma unroll
        for (int r = 0; r < 8; ++r) {
            float v = m_r[r];
#pragma unroll
            for (int t = 0; t < 4; ++t) v = fmaxf(v, s[t][r]);
            v = fmaxf(v, __shfl_xor(v, 8));
            v = fmaxf(v, __shfl_xor(v, 4));
            v = fmaxf(v, __shfl_xor(v, 2));
            v = fmaxf(v, __shfl_xor(v, 1));
            alpha[r] = exp2f((m_r[r] - v) * LOG2E);
            m_r[r]   = v;
            float rs = 0.0f;
#pragma unroll
            for (int t = 0; t < 4; ++t) {
                float p = exp2f((s[t][r] - v) * LOG2E);
                s[t][r] = p;
                rs += p;
            }
            rs += __shfl_xor(rs, 8);
            rs += __shfl_xor(rs, 4);
            rs += __shfl_xor(rs, 2);
            rs += __shfl_xor(rs, 1);
            l_r[r] = l_r[r] * alpha[r] + rs;
        }
#pragma unroll
        for (int dt = 0; dt < 16; ++dt)
#pragma unroll
            for (int r = 0; r < 8; ++r) o[dt][r] *= alpha[r];

        // Stage P (bf16) through wave-private LDS: C-layout -> A-layout.
#pragma unroll
        for (int t = 0; t < 4; ++t)
#pragma unroll
            for (int r = 0; r < 8; ++r)
                sPw[(r + 8 * hi16) * BN + t * 16 + l16] = (bf16_t)s[t][r];

        // ---- GEMM 2: O (16 x D) += P (16 x BN) * K (BN x D) ----
        // Flattened (kt2 x dt) loop with one-step operand lookahead.
        {
            v16bf a_cur = loadA(sPw + l16 * BN, 0, hi16);
            v16bf b_cur = loadB(sKt + l16 * BN, 0, hi16);
#pragma unroll
            for (int idx = 0; idx < (BN / 32) * 16; ++idx) {
                const int dt = idx & 15;
                v16bf a_nxt = a_cur;
                v16bf b_nxt = b_cur;
                if (idx + 1 < (BN / 32) * 16) {
                    const int nk = (idx + 1) >> 4;
                    const int nd = (idx + 1) & 15;
                    if (nd == 0) a_nxt = loadA(sPw + l16 * BN, nk * 32, hi16);
                    b_nxt = loadB(sKt + (nd * 16 + l16) * BN, nk * 32, hi16);
                }
                o[dt] = __builtin_amdgcn_wmma_f32_16x16x32_bf16(
                    false, a_cur, false, b_cur, (short)0, o[dt], false, false);
                a_cur = a_nxt;
                b_cur = b_nxt;
            }
        }
    }

    // ---- epilogue: normalize and write (dir 0) / accumulate (dir 1) ----
    float inv[8];
#pragma unroll
    for (int r = 0; r < 8; ++r) inv[r] = 1.0f / l_r[r];
#pragma unroll
    for (int dt = 0; dt < 16; ++dt) {
#pragma unroll
        for (int r = 0; r < 8; ++r) {
            const int m = wave * 16 + r + 8 * hi16;
            const int d = dt * 16 + l16;
            const float v = o[dt][r] * inv[r];
            float* p = Og + (size_t)m * DD + d;
            if (accumulate) *p = *p + v;
            else            *p = v;
        }
    }
}

extern "C" void kernel_launch(void* const* d_in, const int* in_sizes, int n_in,
                              void* d_out, int out_size, void* d_ws, size_t ws_size,
                              hipStream_t stream) {
    (void)n_in; (void)out_size; (void)d_ws; (void)ws_size;
    const float* x1 = (const float*)d_in[0];
    const float* x2 = (const float*)d_in[1];
    float* out = (float*)d_out;

    const int B = in_sizes[0] / (NSEQ * DD);   // 8
    dim3 grid(NSEQ / BM, B);
    dim3 block(NTHREADS);
    const size_t smem =
        (size_t)(BM * DD + BN * DD + DD * BN + NWAVE * 16 * BN) * sizeof(bf16_t)
        + (size_t)BN * DD * sizeof(float);     // fp32 staging buffer

    // out  = attn(x1, x2)
    mca_flash_attn<<<grid, block, smem, stream>>>(x1, x2, out, 0);
    // out += attn(x2, x1)   (same kernel; S^T direction == swapped roles)
    mca_flash_attn<<<grid, block, smem, stream>>>(x2, x1, out, 1);
}
